// GConv_29703993819414
// MI455X (gfx1250) — matrix operand order repaired and checked by reading, not compiled, not run on hardware
//
#include <hip/hip_runtime.h>

#define NN 20000
#define NPAD 20032            // 313 * 64, row padding for direct fragment loads
#define EE 320000
#define IND 128
#define DD 256
#define GG 128
#define NLAYERS 5

typedef __attribute__((ext_vector_type(16))) __bf16        v16bf;
typedef __attribute__((ext_vector_type(8)))  float         v8f;
typedef __attribute__((ext_vector_type(8)))  unsigned int  v8u;

__device__ __forceinline__ unsigned short f2bf(float f) {
  unsigned u = __builtin_bit_cast(unsigned, f);
  u += 0x7FFFu + ((u >> 16) & 1u);   // round-to-nearest-even
  return (unsigned short)(u >> 16);
}
__device__ __forceinline__ unsigned pk2(float a, float b) {
  return (unsigned)f2bf(a) | ((unsigned)f2bf(b) << 16);
}

// ---------------- elementwise helpers ----------------
__global__ void copy_f32_kernel(const float* __restrict__ in, float* __restrict__ out, int n4) {
  int i = blockIdx.x * blockDim.x + threadIdx.x;
  if (i < n4) ((float4*)out)[i] = ((const float4*)in)[i];
}

__global__ void zero_f32_kernel(float* __restrict__ p, int n) {
  int i = blockIdx.x * blockDim.x + threadIdx.x;
  if (i < n) p[i] = 0.0f;
}

__global__ void zero_i32_kernel(int* __restrict__ p, int n) {
  int i = blockIdx.x * blockDim.x + threadIdx.x;
  if (i < n) p[i] = 0;
}

// weight transpose + bf16: wt[n][k] = bf16(w[k][n]);  w is [K][256]
__global__ void wt_bf16_kernel(const float* __restrict__ w, unsigned short* __restrict__ wt, int K) {
  int i = blockIdx.x * blockDim.x + threadIdx.x;   // over 256*K
  if (i >= 256 * K) return;
  int n = i / K, k = i - n * K;
  wt[i] = f2bf(w[(size_t)k * 256 + n]);
}

// ---------------- GIN edge aggregation (L2-resident scatter-add) ----------------
__global__ void scatter_add_kernel(const float* __restrict__ z, const int* __restrict__ src,
                                   const int* __restrict__ dst, float* __restrict__ agg,
                                   int k4, int total, int K) {
  int i = blockIdx.x * blockDim.x + threadIdx.x;
  if (i >= total) return;
  int e = i / k4;
  int q = (i - e * k4) * 4;
  int s = src[e], t = dst[e];
  float4 v = *(const float4*)(z + (size_t)s * K + q);
  float* o = agg + (size_t)t * K + q;
  atomicAdd(o + 0, v.x); atomicAdd(o + 1, v.y);
  atomicAdd(o + 2, v.z); atomicAdd(o + 3, v.w);
}

// ---------------- bf16 WMMA GEMM, fragments direct from L2 (no LDS, no barriers) ---
// out = act(A[MxK] @ WT^T + bias); WT is [256][K] bf16 (pre-transposed weights).
// 256 threads = 8 waves; block tile 64 x 256; wave tile 16 x 128 (8 accumulators).
// B fragments for the whole K-step are staged into registers BEFORE the wmma chain
// so the 16 global_load_b128 issue as one clause and the wmmas drain them with
// partial loadcnt waits instead of s_wait_loadcnt 0 per wmma.
// A_F32: A is f32 (converted to bf16 in-register), else A is bf16.
template <bool A_F32>
__global__ __launch_bounds__(256)
void gemm_wmma_kernel(const void* __restrict__ Ap,
                      const unsigned short* __restrict__ WT,
                      const float* __restrict__ bias,
                      float* __restrict__ outF,
                      unsigned short* __restrict__ outBF,
                      float* __restrict__ stats,     // if non-null: atomically add col sums/sumsq
                      int M, int K, int doRelu) {
  const int tid    = threadIdx.x;
  const int wave   = tid >> 5;
  const int lane   = tid & 31;
  const int lm     = lane & 15;
  const int lh     = lane >> 4;
  const int row0   = blockIdx.x * 64;
  const int mt     = wave >> 1;            // 0..3 : M sub-tile
  const int ntBase = (wave & 1) * 8;       // 0 or 8 : N sub-tile base
  const int arow   = row0 + mt * 16 + lm;  // A layout: lane lm = row; lh selects K half

  const float*          Af = (const float*)Ap;
  const unsigned short* Ab = (const unsigned short*)Ap;

  v8f acc[8] = {};

  for (int k0 = 0; k0 < K; k0 += 32) {
    // ---- stage all 8 B fragments for this K-step (16 x global_load_b128) ----
    v8u bu[8];
    #pragma unroll
    for (int j = 0; j < 8; ++j) {
      const unsigned short* br =
          WT + (size_t)((ntBase + j) * 16 + lm) * K + k0 + lh * 8;
      uint4 b0 = *(const uint4*)(br);
      uint4 b1 = *(const uint4*)(br + 16);
      bu[j][0] = b0.x; bu[j][1] = b0.y; bu[j][2] = b0.z; bu[j][3] = b0.w;
      bu[j][4] = b1.x; bu[j][5] = b1.y; bu[j][6] = b1.z; bu[j][7] = b1.w;
    }

    // ---- A fragment: 16-bit A 16x32 layout; per lane two contiguous 16B K-chunks ----
    // (f32->bf16 packing VALU executes in the shadow of the B loads)
    v8u au;
    if (A_F32) {
      const float* ar = Af + (size_t)arow * K + k0 + lh * 8;
      float4 f0 = *(const float4*)(ar);
      float4 f1 = *(const float4*)(ar + 4);
      float4 f2 = *(const float4*)(ar + 16);
      float4 f3 = *(const float4*)(ar + 20);
      if (k0 + 32 < K) __builtin_prefetch((const void*)(ar + 32), 0, 3);
      au[0] = pk2(f0.x, f0.y); au[1] = pk2(f0.z, f0.w);
      au[2] = pk2(f1.x, f1.y); au[3] = pk2(f1.z, f1.w);
      au[4] = pk2(f2.x, f2.y); au[5] = pk2(f2.z, f2.w);
      au[6] = pk2(f3.x, f3.y); au[7] = pk2(f3.z, f3.w);
    } else {
      const unsigned short* ar = Ab + (size_t)arow * K + k0 + lh * 8;
      uint4 u0 = *(const uint4*)(ar);
      uint4 u1 = *(const uint4*)(ar + 16);
      if (k0 + 32 < K) __builtin_prefetch((const void*)(ar + 32), 0, 3);
      au[0] = u0.x; au[1] = u0.y; au[2] = u0.z; au[3] = u0.w;
      au[4] = u1.x; au[5] = u1.y; au[6] = u1.z; au[7] = u1.w;
    }
    v16bf afrag = __builtin_bit_cast(v16bf, au);

    // ---- 8 back-to-back wmmas draining the staged fragments ----
    #pragma unroll
    for (int j = 0; j < 8; ++j) {
      v16bf bfrag = __builtin_bit_cast(v16bf, bu[j]);
      acc[j] = __builtin_amdgcn_wmma_f32_16x16x32_bf16(
          /*neg_a=*/false, afrag, /*neg_b=*/false, bfrag,
          /*c_mod=*/(short)0, acc[j], /*reuse_a=*/false, /*reuse_b=*/false);
    }
  }

  // epilogue: C/D layout — VGPR r holds row (r + 8*lh), column = lm.
  // Optionally fuse BatchNorm partial stats (per-column sum / sumsq).
  #pragma unroll
  for (int j = 0; j < 8; ++j) {
    int col = (ntBase + j) * 16 + lm;
    float bv = bias[col];
    float s = 0.0f, s2 = 0.0f;
    #pragma unroll
    for (int r = 0; r < 8; ++r) {
      int grow = row0 + mt * 16 + (lh << 3) + r;
      if (grow < M) {
        float v = acc[j][r] + bv;
        if (doRelu) v = fmaxf(v, 0.0f);
        if (outF)  outF[(size_t)grow * 256 + col]  = v;
        if (outBF) outBF[(size_t)grow * 256 + col] = f2bf(v);
        s += v; s2 += v * v;
      }
    }
    if (stats) {
      atomicAdd(&stats[col], s);
      atomicAdd(&stats[256 + col], s2);
    }
  }
}

// ---------------- BatchNorm apply (+ReLU), dual-write z and next layer's agg init ----
__global__ void bn_apply_kernel(const float* __restrict__ h, const float* __restrict__ stats,
                                const float* __restrict__ gamma, const float* __restrict__ beta,
                                float* __restrict__ out, float* __restrict__ aggInit, int M) {
  int i = blockIdx.x * blockDim.x + threadIdx.x;
  if (i >= M * 256) return;
  int d = i & 255;
  float mu  = stats[d] * (1.0f / (float)M);
  float var = stats[256 + d] * (1.0f / (float)M) - mu * mu;
  float inv = rsqrtf(var + 1e-5f);
  float v = (h[i] - mu) * inv * gamma[d] + beta[d];
  v = fmaxf(v, 0.0f);
  out[i] = v;
  if (aggInit) aggInit[i] = v;
}

// ---------------- graph mean pooling ----------------
__global__ void count_kernel(const int* __restrict__ batch, int* __restrict__ cnt, int n) {
  int i = blockIdx.x * blockDim.x + threadIdx.x;
  if (i < n) atomicAdd(&cnt[batch[i]], 1);
}

__global__ void pool_sum_kernel(const float* __restrict__ z, const int* __restrict__ batch,
                                float* __restrict__ sums, int n) {
  int i = blockIdx.x * blockDim.x + threadIdx.x;   // over n * 64 (4 feats each)
  if (i >= n * 64) return;
  int r = i >> 6;
  int q = (i & 63) * 4;
  int g = batch[r];
  float4 v = *(const float4*)(z + (size_t)r * 256 + q);
  float* o = sums + (size_t)g * 256 + q;
  atomicAdd(o + 0, v.x); atomicAdd(o + 1, v.y);
  atomicAdd(o + 2, v.z); atomicAdd(o + 3, v.w);
}

__global__ void pool_div_kernel(const float* __restrict__ sums, const int* __restrict__ cnt,
                                float* __restrict__ out) {
  int i = blockIdx.x * blockDim.x + threadIdx.x;
  if (i >= GG * 256) return;
  int g = i >> 8;
  out[i] = sums[i] / fmaxf((float)cnt[g], 1.0f);
}

// ---------------- launcher ----------------
extern "C" void kernel_launch(void* const* d_in, const int* in_sizes, int n_in,
                              void* d_out, int out_size, void* d_ws, size_t ws_size,
                              hipStream_t stream) {
  (void)in_sizes; (void)n_in; (void)out_size; (void)ws_size;

  const float* x     = (const float*)d_in[0];
  const int*   ei    = (const int*)d_in[1];
  const int*   src   = ei;
  const int*   dst   = ei + EE;
  const int*   batch = (const int*)d_in[2];
  const float* w1_0  = (const float*)d_in[3];
  const float* b1_0  = (const float*)d_in[4];
  const float* w2_0  = (const float*)d_in[5];
  const float* b2_0  = (const float*)d_in[6];
  const float* g0    = (const float*)d_in[7];
  const float* be0   = (const float*)d_in[8];
  const float* W1    = (const float*)d_in[9];
  const float* B1    = (const float*)d_in[10];
  const float* W2    = (const float*)d_in[11];
  const float* B2    = (const float*)d_in[12];
  const float* Gm    = (const float*)d_in[13];
  const float* BE    = (const float*)d_in[14];

  char* ws = (char*)d_ws;
  auto wsalloc = [&](size_t bytes) -> void* {
    void* p = (void*)ws;
    ws += (bytes + 255) & ~(size_t)255;
    return p;
  };
  float*          agg  = (float*)         wsalloc((size_t)NPAD * DD * 4);
  float*          h    = (float*)         wsalloc((size_t)NPAD * DD * 4);
  float*          zb   = (float*)         wsalloc((size_t)NN * DD * 4);
  unsigned short* hbf  = (unsigned short*)wsalloc((size_t)NPAD * DD * 2);
  unsigned short* w1t  = (unsigned short*)wsalloc((size_t)DD * DD * 2);
  unsigned short* w2t  = (unsigned short*)wsalloc((size_t)DD * DD * 2);
  float*          stats= (float*)         wsalloc(512 * 4);
  float*          psum = (float*)         wsalloc((size_t)GG * DD * 4);
  int*            pcnt = (int*)           wsalloc((size_t)GG * 4);

  float* outNode  = (float*)d_out;
  float* outGraph = outNode + (size_t)NN * DD;

  // layer 0 agg init: agg = x (later layers: bn_apply dual-writes agg)
  copy_f32_kernel<<<(NN * IND / 4 + 255) / 256, 256, 0, stream>>>(x, agg, NN * IND / 4);

  const float* zin = x;
  int K = IND;
  const int gblocks = (NN + 63) / 64;     // 313 -> covers exactly NPAD rows
  for (int l = 0; l < NLAYERS; ++l) {
    const float* w1 = (l == 0) ? w1_0 : W1 + (size_t)(l - 1) * DD * DD;
    const float* b1 = (l == 0) ? b1_0 : B1 + (size_t)(l - 1) * DD;
    const float* w2 = (l == 0) ? w2_0 : W2 + (size_t)(l - 1) * DD * DD;
    const float* b2 = (l == 0) ? b2_0 : B2 + (size_t)(l - 1) * DD;
    const float* gm = (l == 0) ? g0   : Gm + (size_t)(l - 1) * DD;
    const float* be = (l == 0) ? be0  : BE + (size_t)(l - 1) * DD;

    // agg += scatter(z[src] -> dst)   (agg pre-initialized with z)
    int tot = EE * (K / 4);
    scatter_add_kernel<<<(tot + 255) / 256, 256, 0, stream>>>(zin, src, dst, agg, K / 4, tot, K);

    // transpose+convert weights (tiny, L2-resident)
    wt_bf16_kernel<<<(256 * K + 255) / 256, 256, 0, stream>>>(w1, w1t, K);
    wt_bf16_kernel<<<(256 * DD + 255) / 256, 256, 0, stream>>>(w2, w2t, DD);
    zero_f32_kernel<<<2, 256, 0, stream>>>(stats, 512);

    // MLP: h = (relu(agg @ W1 + b1)) @ W2 + b2 ; GEMM2 fuses BN stats
    gemm_wmma_kernel<true ><<<gblocks, 256, 0, stream>>>(agg, w1t, b1, nullptr, hbf, nullptr, NN, K, 1);
    gemm_wmma_kernel<false><<<gblocks, 256, 0, stream>>>(hbf, w2t, b2, h, nullptr, stats, NN, DD, 0);

    // BN + ReLU; dual-write next layer's agg init
    float* zout    = (l == NLAYERS - 1) ? outNode : zb;
    float* aggNext = (l == NLAYERS - 1) ? nullptr : agg;
    bn_apply_kernel<<<(NN * DD + 255) / 256, 256, 0, stream>>>(h, stats, gm, be, zout, aggNext, NN);

    zin = zout;
    K = DD;
  }

  // graph mean pooling
  zero_f32_kernel<<<(GG * DD + 255) / 256, 256, 0, stream>>>(psum, GG * DD);
  zero_i32_kernel<<<1, GG, 0, stream>>>(pcnt, GG);
  count_kernel<<<(NN + 255) / 256, 256, 0, stream>>>(batch, pcnt, NN);
  pool_sum_kernel<<<(NN * 64 + 255) / 256, 256, 0, stream>>>(outNode, batch, psum, NN);
  pool_div_kernel<<<(GG * DD + 255) / 256, 256, 0, stream>>>(psum, pcnt, outGraph);
}
